// ProteinFeatures_52347061404121
// MI455X (gfx1250) — matrix operand order repaired and checked by reading, not compiled, not run on hardware
//
#include <hip/hip_runtime.h>
#include <hip/hip_fp16.h>

typedef __attribute__((ext_vector_type(16))) _Float16 v16h;
typedef __attribute__((ext_vector_type(8)))  float    v8f;

#define B_DIM 4
#define L_DIM 2048
#define TOPK 30
#define NRBF 16
#define KFEAT 416          // 16 pos + 25*16 rbf
#define NOUT 128
#define TOTAL_RES (B_DIM * L_DIM)
#define TOTAL_E (TOTAL_RES * TOPK)   // 245760
#define EPB 128            // edges per block in fused gemm kernel
#define MTILES 2           // 16-row M tiles per wave
#define NPAIR 25

// atom-pair table: p=0 -> D_neighbors (precomputed); p=1..24 -> (center atom, neighbor atom)
__constant__ int PA[NPAIR] = {1, 0,2,3,4, 1,1,1,1, 0,0,0, 4,4, 3, 0,2,3,4, 2,3,4, 2,3, 2};
__constant__ int PB[NPAIR] = {1, 0,2,3,4, 0,2,3,4, 2,3,4, 2,3, 2, 1,1,1,1, 0,0,0, 4,4, 3};

// ---------------------------------------------------------------------------
// Kernel 1: derive N, Ca, C, O, Cb for each residue -> atoms[res][5][3]
// ---------------------------------------------------------------------------
__global__ void pf_atoms_kernel(const float* __restrict__ X, float* __restrict__ atoms) {
    int idx = blockIdx.x * blockDim.x + threadIdx.x;
    if (idx >= TOTAL_RES) return;
    const float* x = X + (size_t)idx * 12;
    float Nx = x[0],  Ny = x[1],  Nz = x[2];
    float Ax = x[3],  Ay = x[4],  Az = x[5];   // Ca
    float Cx = x[6],  Cy = x[7],  Cz = x[8];
    float Ox = x[9],  Oy = x[10], Oz = x[11];
    float bx = Ax - Nx, by = Ay - Ny, bz = Az - Nz;
    float cx = Cx - Ax, cy = Cy - Ay, cz = Cz - Az;
    float ax = by * cz - bz * cy;
    float ay = bz * cx - bx * cz;
    float az = bx * cy - by * cx;
    float Bx = -0.58273431f * ax + 0.56802827f * bx - 0.54067466f * cx + Ax;
    float By = -0.58273431f * ay + 0.56802827f * by - 0.54067466f * cy + Ay;
    float Bz = -0.58273431f * az + 0.56802827f * bz - 0.54067466f * cz + Az;
    float* o = atoms + (size_t)idx * 15;
    o[0] = Nx; o[1] = Ny; o[2] = Nz;
    o[3] = Ax; o[4] = Ay; o[5] = Az;
    o[6] = Cx; o[7] = Cy; o[8] = Cz;
    o[9] = Ox; o[10] = Oy; o[11] = Oz;
    o[12] = Bx; o[13] = By; o[14] = Bz;
}

// ---------------------------------------------------------------------------
// Kernel 2: per-row masked distances + top-30 (ascending) selection.
// One 256-thread block per (b, i) row.  Packed u64 keys (valbits<<32 | j)
// reproduce top_k's value-then-index ordering for non-negative distances.
// ---------------------------------------------------------------------------
__global__ void pf_topk_kernel(const float* __restrict__ atoms,
                               const float* __restrict__ mask,
                               float* __restrict__ Dnb,
                               int* __restrict__ Eidx,
                               int* __restrict__ Eidx_out) {
    __shared__ float ds[L_DIM];
    __shared__ float redf[256];
    __shared__ unsigned long long redu[256];

    const int row = blockIdx.x;            // b * L + i
    const int b = row / L_DIM;
    const int tid = threadIdx.x;
    const float* CaB = atoms + (size_t)b * L_DIM * 15;
    const float* mB = mask + (size_t)b * L_DIM;

    const float cax = CaB[(size_t)(row % L_DIM) * 15 + 3];
    const float cay = CaB[(size_t)(row % L_DIM) * 15 + 4];
    const float caz = CaB[(size_t)(row % L_DIM) * 15 + 5];
    const float mi = mask[row];

    float lmax = 0.0f;
    for (int j = tid; j < L_DIM; j += 256) {
        float dx = cax - CaB[(size_t)j * 15 + 3];
        float dy = cay - CaB[(size_t)j * 15 + 4];
        float dz = caz - CaB[(size_t)j * 15 + 5];
        float dist = sqrtf(dx * dx + dy * dy + dz * dz + 1e-6f);
        float dv = (mi * mB[j]) * dist;
        ds[j] = dv;
        lmax = fmaxf(lmax, dv);
    }
    redf[tid] = lmax;
    __syncthreads();
    for (int s = 128; s > 0; s >>= 1) {
        if (tid < s) redf[tid] = fmaxf(redf[tid], redf[tid + s]);
        __syncthreads();
    }
    const float Dmax = redf[0];
    __syncthreads();
    for (int j = tid; j < L_DIM; j += 256) {
        float m2 = mi * mB[j];
        ds[j] += (1.0f - m2) * Dmax;     // D_adj
    }
    __syncthreads();

    for (int k = 0; k < TOPK; ++k) {
        unsigned long long best = ~0ull;
        for (int j = tid; j < L_DIM; j += 256) {
            unsigned long long key =
                ((unsigned long long)__float_as_uint(ds[j]) << 32) | (unsigned)j;
            best = (key < best) ? key : best;
        }
        redu[tid] = best;
        __syncthreads();
        for (int s = 128; s > 0; s >>= 1) {
            if (tid < s) redu[tid] = (redu[tid + s] < redu[tid]) ? redu[tid + s] : redu[tid];
            __syncthreads();
        }
        if (tid == 0) {
            unsigned long long w = redu[0];
            int jmin = (int)(w & 0xffffffffull);
            float dv = __uint_as_float((unsigned)(w >> 32));
            size_t o = (size_t)row * TOPK + k;
            Dnb[o] = dv;
            Eidx[o] = jmin;
            Eidx_out[o] = jmin;
            ds[jmin] = __uint_as_float(0x7f800000u);  // +inf: exclude
        }
        __syncthreads();
    }
}

// ---------------------------------------------------------------------------
// Kernel 3 (fused): per-block build 128 edges x 416 f16 features in LDS,
// WMMA-GEMM against W (staged col-major in LDS), LayerNorm(128), store.
// Block = 128 threads = 4 waves; wave w owns M rows [w*32, w*32+32)
// as two 16-row tiles so each B-fragment load feeds two WMMAs.
// K-loop rolled (bounds VGPR liveness; no spills).  B fragments are
// manually double-buffered so LDS loads overlap WMMA issue.
// ---------------------------------------------------------------------------
__global__ void __launch_bounds__(128, 1)
pf_edge_gemm_kernel(const float* __restrict__ atoms,
                    const float* __restrict__ Dnb,
                    const int* __restrict__ Eidx,
                    const int* __restrict__ ridx,
                    const int* __restrict__ chain,
                    const float* __restrict__ pos_w,
                    const float* __restrict__ pos_b,
                    const float* __restrict__ edge_w,
                    const float* __restrict__ ln_w,
                    const float* __restrict__ ln_b,
                    float* __restrict__ Eout) {
    extern __shared__ char smem[];
    _Float16* Wt = (_Float16*)smem;                                // [128][416] col-major weights
    _Float16* Fs = (_Float16*)(smem + (size_t)NOUT * KFEAT * 2);   // [128][416] edge features
    float* ED = (float*)(smem + (size_t)(NOUT + EPB) * KFEAT * 2); // [128][25] distances
    int* DP = (int*)(ED + EPB * NPAIR);                            // [128] positional index

    const int tid = threadIdx.x;
    const int e0 = blockIdx.x * EPB;

    // Phase 1: stage W transposed: Wt[n][k] = edge_w[k][n]  (f16)
    for (int k = 0; k < KFEAT; ++k)
        Wt[(size_t)tid * KFEAT + k] = (_Float16)edge_w[(size_t)k * NOUT + tid];

    // Phase 2a: per-edge pair distances + positional bucket
    for (int idx = tid; idx < EPB * NPAIR; idx += 128) {
        int el = idx / NPAIR;
        int p = idx - el * NPAIR;
        int e = e0 + el;
        float dist = 0.0f;
        if (e < TOTAL_E) {
            if (p == 0) {
                dist = Dnb[e];
            } else {
                int bi = e / (L_DIM * TOPK);
                int rem = e - bi * (L_DIM * TOPK);
                int i = rem / TOPK;
                int j = Eidx[e];
                const float* pi = atoms + (((size_t)bi * L_DIM + i) * 5 + PA[p]) * 3;
                const float* pj = atoms + (((size_t)bi * L_DIM + j) * 5 + PB[p]) * 3;
                float dx = pi[0] - pj[0], dy = pi[1] - pj[1], dz = pi[2] - pj[2];
                dist = sqrtf(dx * dx + dy * dy + dz * dz + 1e-6f);
            }
        }
        ED[el * NPAIR + p] = dist;
    }
    {
        int e = e0 + tid;
        int d = 65;
        if (e < TOTAL_E) {
            int bi = e / (L_DIM * TOPK);
            int rem = e - bi * (L_DIM * TOPK);
            int i = rem / TOPK;
            int j = Eidx[e];
            int off = ridx[bi * L_DIM + i] - ridx[bi * L_DIM + j];
            int same = (chain[bi * L_DIM + i] == chain[bi * L_DIM + j]);
            int dc = off + 32;
            dc = dc < 0 ? 0 : (dc > 64 ? 64 : dc);
            d = same ? dc : 65;
        }
        DP[tid] = d;
    }
    __syncthreads();

    // Phase 2b: materialize f16 features: [pos(16) | 25 x RBF(16)]
    for (int idx = tid; idx < EPB * KFEAT; idx += 128) {
        int el = idx / KFEAT;
        int f = idx - el * KFEAT;
        float v;
        if (f < 16) {
            v = pos_w[DP[el] * 16 + f] + pos_b[f];
        } else {
            int q = f - 16;
            int p = q >> 4;
            int r = q & 15;
            float dloc = ED[el * NPAIR + p];
            float mu = 2.0f + (20.0f / 15.0f) * (float)r;
            float t = (dloc - mu) * (1.0f / 1.25f);
            v = __expf(-t * t);
        }
        Fs[(size_t)el * KFEAT + f] = (_Float16)v;
    }
    __syncthreads();

    // Phase 3: WMMA GEMM.  K = 416 = 13 x 32; 2 M-tiles x 8 N-tiles per wave.
    const int wid = tid >> 5;
    const int lane = tid & 31;
    const int half = lane >> 4;
    const int lm = lane & 15;

    v8f acc[MTILES][8];
#pragma unroll
    for (int mt = 0; mt < MTILES; ++mt)
#pragma unroll
        for (int t = 0; t < 8; ++t)
            acc[mt][t] = (v8f){0.f, 0.f, 0.f, 0.f, 0.f, 0.f, 0.f, 0.f};

    union Frag { v16h h; uint4 q[2]; };

    const _Float16* arowBase = Fs + (size_t)(wid * 32 + lm) * KFEAT + (half << 4);
    const _Float16* browBase = Wt + (size_t)lm * KFEAT + (half << 4);

    Frag bf[2];
    // prime B pipeline: (kt=0, nt=0)
    bf[0].q[0] = *(const uint4*)(browBase);
    bf[0].q[1] = *(const uint4*)(browBase + 8);

#pragma unroll 1   // keep rolled: bounds VGPR liveness, no accumulator spills
    for (int kt = 0; kt < 13; ++kt) {
        const int k0 = kt * 32;
        // A fragments (16x32 f16): per-lane two contiguous 16B runs -> ds_load_b128
        Frag af[MTILES];
#pragma unroll
        for (int mt = 0; mt < MTILES; ++mt) {
            const _Float16* arow = arowBase + (size_t)(mt * 16) * KFEAT + k0;
            af[mt].q[0] = *(const uint4*)(arow);
            af[mt].q[1] = *(const uint4*)(arow + 8);
        }
#pragma unroll
        for (int nt = 0; nt < 8; ++nt) {
            const int cur = nt & 1;
            const int nxt = cur ^ 1;
            // prefetch next B fragment (next nt, or nt=0 of next kt)
            if (nt < 7) {
                const _Float16* brow = browBase + (size_t)((nt + 1) * 16) * KFEAT + k0;
                bf[nxt].q[0] = *(const uint4*)(brow);
                bf[nxt].q[1] = *(const uint4*)(brow + 8);
            } else if (kt < 12) {
                const _Float16* brow = browBase + (k0 + 32);
                bf[nxt].q[0] = *(const uint4*)(brow);
                bf[nxt].q[1] = *(const uint4*)(brow + 8);
            }
#pragma unroll
            for (int mt = 0; mt < MTILES; ++mt)
                acc[mt][nt] = __builtin_amdgcn_wmma_f32_16x16x32_f16(
                    false, af[mt].h, false, bf[cur].h, (short)0, acc[mt][nt], false, false);
        }
    }

    // Phase 4: LayerNorm over 128 cols of each row + coalesced store.
    float lnw[8], lnb[8];
#pragma unroll
    for (int t = 0; t < 8; ++t) {
        lnw[t] = ln_w[t * 16 + lm];
        lnb[t] = ln_b[t * 16 + lm];
    }
#pragma unroll
    for (int mt = 0; mt < MTILES; ++mt) {
#pragma unroll
        for (int r = 0; r < 8; ++r) {
            float s = 0.f, sq = 0.f;
#pragma unroll
            for (int t = 0; t < 8; ++t) {
                float x = acc[mt][t][r];
                s += x;
                sq += x * x;
            }
            // rows live in disjoint 16-lane halves: xor masks 1..8 stay in-half
            for (int m = 1; m <= 8; m <<= 1) {
                s += __shfl_xor(s, m, 32);
                sq += __shfl_xor(sq, m, 32);
            }
            float mean = s * (1.0f / 128.0f);
            float var = sq * (1.0f / 128.0f) - mean * mean;
            float inv = rsqrtf(var + 1e-5f);
            int erow = e0 + wid * 32 + mt * 16 + r + 8 * half;
            if (erow < TOTAL_E) {
                float* orow = Eout + (size_t)erow * NOUT;
#pragma unroll
                for (int t = 0; t < 8; ++t)
                    orow[t * 16 + lm] = (acc[mt][t][r] - mean) * inv * lnw[t] + lnb[t];
            }
        }
    }
}

// ---------------------------------------------------------------------------
extern "C" void kernel_launch(void* const* d_in, const int* in_sizes, int n_in,
                              void* d_out, int out_size, void* d_ws, size_t ws_size,
                              hipStream_t stream) {
    const float* X      = (const float*)d_in[0];
    const float* mask   = (const float*)d_in[1];
    const int*   ridx   = (const int*)d_in[2];
    const int*   chain  = (const int*)d_in[3];
    const float* pos_w  = (const float*)d_in[4];
    const float* pos_b  = (const float*)d_in[5];
    const float* edge_w = (const float*)d_in[6];
    const float* ln_w   = (const float*)d_in[7];
    const float* ln_b   = (const float*)d_in[8];

    float* E_out = (float*)d_out;
    int* Eidx_out = (int*)(E_out + (size_t)TOTAL_E * NOUT);

    // workspace layout: atoms | D_neighbors | E_idx  (~2.5 MB)
    float* atoms = (float*)d_ws;
    float* Dnb = atoms + (size_t)TOTAL_RES * 15;
    int* Eidx = (int*)(Dnb + TOTAL_E);

    pf_atoms_kernel<<<(TOTAL_RES + 255) / 256, 256, 0, stream>>>(X, atoms);
    pf_topk_kernel<<<TOTAL_RES, 256, 0, stream>>>(atoms, mask, Dnb, Eidx, Eidx_out);

    size_t lds = (size_t)(NOUT + EPB) * KFEAT * 2   // Wt + Fs (f16)
               + (size_t)EPB * NPAIR * 4            // ED
               + (size_t)EPB * 4;                   // DP
    pf_edge_gemm_kernel<<<TOTAL_E / EPB, 128, lds, stream>>>(
        atoms, Dnb, Eidx, ridx, chain, pos_w, pos_b, edge_w, ln_w, ln_b, E_out);
}
